// ControllerLatent_36266703847975
// MI455X (gfx1250) — matrix operand order repaired and checked by reading, not compiled, not run on hardware
//
#include <hip/hip_runtime.h>
#include <hip/hip_bf16.h>

// ---------------------------------------------------------------------------
// GRU over 1024 serialized steps, batch 64, D=H=1024.
// Persistent-scan kernel: 16 blocks x 128 threads = 64 waves; wave w owns
// 16 hidden columns. Split grid barrier (arrive -> overlap -> wait): only the
// h-projection (384 v_wmma_f32_16x16x32_bf16 per wave) is on the serial
// critical path; the x-projection of step t+1 and the output stores execute
// between arrive and wait, hidden behind the barrier round-trip.
// ---------------------------------------------------------------------------

typedef __bf16 bf16_t;
typedef __attribute__((ext_vector_type(16))) __bf16 v16bf;
typedef __attribute__((ext_vector_type(8)))  __bf16 v8bf;
typedef __attribute__((ext_vector_type(8)))  float  v8f;

#define GRU_H     1024
#define GRU_NB    64          // batch
#define GRU_D     1024
#define GRU_STEPS 1024        // T*L
#define NBLK      16
#define TPB       128

// ---------------- prep kernels ----------------

// x (L=8, N=64, T=128, D=1024) fp32  ->  xs (T*L, N, D) bf16, time-major
__global__ void k_convert_x(const float* __restrict__ x, bf16_t* __restrict__ xs) {
  const size_t total = (size_t)GRU_STEPS * GRU_NB * GRU_D;
  for (size_t o = (size_t)blockIdx.x * blockDim.x + threadIdx.x; o < total;
       o += (size_t)gridDim.x * blockDim.x) {
    size_t d  = o & (GRU_D - 1);
    size_t sn = o >> 10;
    size_t n  = sn & (GRU_NB - 1);
    size_t s  = sn >> 6;          // s = t*8 + l
    size_t t  = s >> 3;
    size_t l  = s & 7;
    size_t in = (((l * GRU_NB + n) * 128 + t) << 10) + d;
    xs[o] = (bf16_t)x[in];
  }
}

__global__ void k_convert_w(const float* __restrict__ w, bf16_t* __restrict__ wb, int count) {
  for (int i = blockIdx.x * blockDim.x + threadIdx.x; i < count;
       i += gridDim.x * blockDim.x)
    wb[i] = (bf16_t)w[i];
}

// h buffer 0 <- rnn_state (fp32 + bf16); zero barrier counter
__global__ void k_init_h(const float* __restrict__ h0, float* __restrict__ hf,
                         bf16_t* __restrict__ hb, unsigned int* __restrict__ cnt) {
  const int total = GRU_NB * GRU_H;
  for (int i = blockIdx.x * blockDim.x + threadIdx.x; i < total;
       i += gridDim.x * blockDim.x) {
    float v = h0[i];
    hf[i] = v;
    hb[i] = (bf16_t)v;
  }
  if (blockIdx.x == 0 && threadIdx.x == 0) *cnt = 0u;
}

// ---------------- persistent GRU scan ----------------

__device__ __forceinline__ v8f splat8(float v) {
  v8f r;
#pragma unroll
  for (int i = 0; i < 8; ++i) r[i] = v;
  return r;
}

__device__ __forceinline__ float fast_sigmoid(float x) {
  // 1/(1+e^-x) with v_exp_f32 + v_rcp_f32 (saturating gate; approx rcp is fine)
  return __builtin_amdgcn_rcpf(1.0f + __expf(-x));
}

__device__ __forceinline__ float fast_tanh(float x) {
  // 1 - 2/(e^{2x}+1); saturates correctly for |x| large
  float e = __expf(2.0f * x);
  return 1.0f - 2.0f * __builtin_amdgcn_rcpf(e + 1.0f);
}

__device__ __forceinline__ v16bf load_a_frag(const bf16_t* p) {
  // A 16x32 bf16 layout: this lane's row, chunk0 = K[base..base+7],
  // chunk1 = K[base+8..base+15]  (lane-half K offset folded into p by caller)
  union { v16bf v; v8bf h[2]; } u;
  u.h[0] = *(const v8bf*)(p);
  u.h[1] = *(const v8bf*)(p + 8);
  return u.v;
}

// x-projection of one step: accx[mt][g] = bias_g + x_t-tiles @ W_ih rows
__device__ __forceinline__ void x_project(v8f (&accx)[4][3],
                                          const bf16_t* __restrict__ xrow,
                                          const bf16_t* const (&wih_row)[3],
                                          int colL, int half,
                                          float biasr, float biasz, float biasni) {
#pragma unroll
  for (int mt = 0; mt < 4; ++mt) {
    accx[mt][0] = splat8(biasr);
    accx[mt][1] = splat8(biasz);
    accx[mt][2] = splat8(biasni);
  }
#pragma unroll 2
  for (int kk = 0; kk < GRU_D; kk += 32) {
    v16bf ax[4];
#pragma unroll
    for (int mt = 0; mt < 4; ++mt)
      ax[mt] = load_a_frag(xrow + (size_t)(mt * 16 + colL) * GRU_D + kk + half * 16);
#pragma unroll
    for (int g = 0; g < 3; ++g) {
      v16bf b = *(const v16bf*)(wih_row[g] + kk);
#pragma unroll
      for (int mt = 0; mt < 4; ++mt)
        accx[mt][g] = __builtin_amdgcn_wmma_f32_16x16x32_bf16(
            false, ax[mt], false, b, (short)0, accx[mt][g], false, false);
    }
  }
}

__global__ __launch_bounds__(TPB, 1)
void k_gru_scan(const bf16_t* __restrict__ xs,    // (1024, 64, 1024) bf16
                const bf16_t* __restrict__ wih,   // (3072, 1024) bf16 row-major
                const bf16_t* __restrict__ whh,   // (3072, 1024) bf16 row-major
                const float*  __restrict__ b_ih,  // (3072,)
                const float*  __restrict__ b_hh,  // (3072,)
                float*        __restrict__ hf,    // 2 x (64,1024) fp32
                bf16_t*       __restrict__ hb,    // 2 x (64,1024) bf16
                float*        __restrict__ out,   // (64, 1024, 1024)
                float*        __restrict__ hlast, // (64, 1024)
                unsigned int* __restrict__ cnt)
{
  const int lane = threadIdx.x & 31;
  const int wid  = (blockIdx.x * TPB + threadIdx.x) >> 5;  // 0..63
  const int colL = lane & 15;       // column within tile / A-row within tile
  const int half = lane >> 4;       // K-half select for WMMA operand layout
  const int col  = wid * 16 + colL; // hidden column this lane produces

  // Per-lane bias terms (uniform across both lane halves with same colL)
  const float biasr  = b_ih[col] + b_hh[col];
  const float biasz  = b_ih[GRU_H + col] + b_hh[GRU_H + col];
  const float biasni = b_ih[2 * GRU_H + col];
  const float biasnh = b_hh[2 * GRU_H + col];

  // B-operand row bases: W row (g*H + col), contiguous along K
  const bf16_t* wih_row[3];
  const bf16_t* whh_row[3];
#pragma unroll
  for (int g = 0; g < 3; ++g) {
    wih_row[g] = wih + ((size_t)(g * GRU_H + col)) * GRU_D + half * 16;
    whh_row[g] = whh + ((size_t)(g * GRU_H + col)) * GRU_H + half * 16;
  }

  // Prologue: x-projection for step 0 (fully parallel, before any sync)
  v8f accx[4][3];
  x_project(accx, xs, wih_row, colL, half, biasr, biasz, biasni);

  for (int step = 0; step < GRU_STEPS; ++step) {
    const bf16_t* hread_b = hb + (size_t)(step & 1) * GRU_NB * GRU_H;
    const float*  hread_f = hf + (size_t)(step & 1) * GRU_NB * GRU_H;
    bf16_t*       hw_b    = hb + (size_t)((step + 1) & 1) * GRU_NB * GRU_H;
    float*        hw_f    = hf + (size_t)((step + 1) & 1) * GRU_NB * GRU_H;

    // Move x-part into combined accumulators; h_n part seeded with its bias.
    v8f acc[4][4];
#pragma unroll
    for (int mt = 0; mt < 4; ++mt) {
      acc[mt][0] = accx[mt][0];
      acc[mt][1] = accx[mt][1];
      acc[mt][2] = accx[mt][2];
      acc[mt][3] = splat8(biasnh);
    }

    // ---- serialized part: h_t @ W_hh^T (384 WMMAs / wave) ----
#pragma unroll 2
    for (int kk = 0; kk < GRU_H; kk += 32) {
      v16bf ah[4];
#pragma unroll
      for (int mt = 0; mt < 4; ++mt)
        ah[mt] = load_a_frag(hread_b + (size_t)(mt * 16 + colL) * GRU_H + kk + half * 16);
#pragma unroll
      for (int g = 0; g < 3; ++g) {
        v16bf b = *(const v16bf*)(whh_row[g] + kk);
        const int ai = (g < 2) ? g : 3;
#pragma unroll
        for (int mt = 0; mt < 4; ++mt)
          acc[mt][ai] = __builtin_amdgcn_wmma_f32_16x16x32_bf16(
              false, ah[mt], false, b, (short)0, acc[mt][ai], false, false);
      }
    }

    // ---- gates + h update (biases already folded into acc) ----
    // C layout: lane holds (m = mt*16 + half*8 + r, n = col) at vgpr r.
    float hnew[4][8];
#pragma unroll
    for (int mt = 0; mt < 4; ++mt) {
#pragma unroll
      for (int r = 0; r < 8; ++r) {
        const int m = mt * 16 + half * 8 + r;
        float fr = fast_sigmoid(acc[mt][0][r]);
        float fz = fast_sigmoid(acc[mt][1][r]);
        float fn = fast_tanh(acc[mt][2][r] + fr * acc[mt][3][r]);
        float hp = hread_f[(size_t)m * GRU_H + col];
        float hn = fz * hp + (1.0f - fz) * fn;
        hnew[mt][r] = hn;
        hw_f[(size_t)m * GRU_H + col] = hn;              // visible before arrive
        hw_b[(size_t)m * GRU_H + col] = (bf16_t)hn;
      }
    }

    // ---- barrier ARRIVE: h writes visible, signal this block done ----
    __threadfence();
    __syncthreads();
    if (threadIdx.x == 0)
      __hip_atomic_fetch_add(cnt, 1u, __ATOMIC_RELEASE, __HIP_MEMORY_SCOPE_AGENT);

    // ---- overlap region: work with no cross-step dependency ----
    // (a) output stores (nobody reads them back)
#pragma unroll
    for (int mt = 0; mt < 4; ++mt)
#pragma unroll
      for (int r = 0; r < 8; ++r) {
        const int m = mt * 16 + half * 8 + r;
        out[((size_t)m * GRU_STEPS + step) * GRU_H + col] = hnew[mt][r];
        if (step == GRU_STEPS - 1)
          hlast[(size_t)m * GRU_H + col] = hnew[mt][r];
      }
    // (b) prefetch x slice for step+2 into L2 (block 0 only; L2 is global)
    if (wid < 2 && step + 2 < GRU_STEPS) {
      const bf16_t* nx = xs + ((size_t)(step + 2) * GRU_NB + (wid * 32 + lane)) * GRU_D;
#pragma unroll
      for (int q = 0; q < 4; ++q)
        __builtin_prefetch(nx + q * 256, 0, 1);   // global_prefetch_b8
    }
    // (c) x-projection for step+1 (384 WMMAs hidden behind barrier latency)
    if (step + 1 < GRU_STEPS)
      x_project(accx, xs + (size_t)(step + 1) * GRU_NB * GRU_D, wih_row,
                colL, half, biasr, biasz, biasni);

    // ---- barrier WAIT: all blocks arrived for this step ----
    if (threadIdx.x == 0) {
      const unsigned target = (unsigned)(step + 1) * NBLK;
      while (__hip_atomic_load(cnt, __ATOMIC_ACQUIRE, __HIP_MEMORY_SCOPE_AGENT) < target)
        __builtin_amdgcn_s_sleep(1);
    }
    __syncthreads();
    __threadfence();
  }
}

// ---------------- host launcher ----------------

extern "C" void kernel_launch(void* const* d_in, const int* in_sizes, int n_in,
                              void* d_out, int out_size, void* d_ws, size_t ws_size,
                              hipStream_t stream) {
  (void)in_sizes; (void)n_in; (void)out_size; (void)ws_size;

  const float* x     = (const float*)d_in[0];  // (8, 64, 128, 1024)
  const float* h0    = (const float*)d_in[1];  // (1, 64, 1024)
  const float* W_ih  = (const float*)d_in[2];  // (3072, 1024)
  const float* W_hh  = (const float*)d_in[3];  // (3072, 1024)
  const float* b_ih  = (const float*)d_in[4];  // (3072,)
  const float* b_hh  = (const float*)d_in[5];  // (3072,)

  float* out   = (float*)d_out;                            // (64, 1024, 1024)
  float* hlast = out + (size_t)GRU_NB * GRU_STEPS * GRU_H; // (1, 64, 1024)

  // workspace carve-up (~148 MB total), 256B aligned chunks
  char* ws = (char*)d_ws;
  size_t off = 0;
  auto carve = [&](size_t bytes) {
    void* p = ws + off;
    off += (bytes + 255) & ~(size_t)255;
    return p;
  };
  bf16_t*       xs   = (bf16_t*)carve((size_t)GRU_STEPS * GRU_NB * GRU_D * 2); // 134 MB
  bf16_t*       wihb = (bf16_t*)carve((size_t)3 * GRU_H * GRU_D * 2);          // 6.3 MB
  bf16_t*       whhb = (bf16_t*)carve((size_t)3 * GRU_H * GRU_H * 2);          // 6.3 MB
  float*        hf   = (float*) carve((size_t)2 * GRU_NB * GRU_H * 4);         // 512 KB
  bf16_t*       hbuf = (bf16_t*)carve((size_t)2 * GRU_NB * GRU_H * 2);         // 256 KB
  unsigned int* cnt  = (unsigned int*)carve(256);

  k_convert_x<<<4096, 256, 0, stream>>>(x, xs);
  k_convert_w<<<1536, 256, 0, stream>>>(W_ih, wihb, 3 * GRU_H * GRU_D);
  k_convert_w<<<1536, 256, 0, stream>>>(W_hh, whhb, 3 * GRU_H * GRU_H);
  k_init_h<<<64, 256, 0, stream>>>(h0, hf, hbuf, cnt);

  k_gru_scan<<<NBLK, TPB, 0, stream>>>(xs, wihb, whhb, b_ih, b_hh,
                                       hf, hbuf, out, hlast, cnt);
}
